// Attention_82944408420965
// MI455X (gfx1250) — matrix operand order repaired and checked by reading, not compiled, not run on hardware
//
#include <hip/hip_runtime.h>
#include <hip/hip_bf16.h>

// ---------------- problem constants ----------------
constexpr int B_   = 2;
constexpr int S_   = 2048;
constexpr int DIM_ = 4096;
constexpr int H_   = 32;
constexpr int D_   = 128;          // head dim
constexpr int AL_  = 10;           // adapter len
constexpr int MT_  = B_ * S_;      // 4096 tokens
constexpr float SCALE_ = 0.08838834764831845f; // 1/sqrt(128)

// ---------------- CDNA5 WMMA types ----------------
typedef __attribute__((ext_vector_type(16))) __bf16 v16bf;
typedef __attribute__((ext_vector_type(8)))  float  v8f;

union Frag16 {            // 16 bf16 = 8 VGPRs (A or B operand)
  v16bf v;
  uint4 q[2];
  unsigned u[8];
  unsigned short h[16];
};
union FragC {             // 16x16 f32 accumulator = 8 VGPRs
  v8f v;
  float f[8];
};

// ---------------- helpers ----------------
__device__ __forceinline__ unsigned short f2bf(float f) {
  union { float f; unsigned u; } v; v.f = f;
  unsigned u = v.u;
  u += 0x7FFFu + ((u >> 16) & 1u);       // round-to-nearest-even
  return (unsigned short)(u >> 16);
}
__device__ __forceinline__ unsigned pack2(float lo, float hi) {
  return (unsigned)f2bf(lo) | ((unsigned)f2bf(hi) << 16);
}
__device__ __forceinline__ float bf2f(unsigned short h) {
  union { unsigned u; float f; } v; v.u = ((unsigned)h) << 16;
  return v.f;
}

// ---------------- f32 -> bf16 convert ----------------
__global__ void cvt_bf16_kernel(const float* __restrict__ src,
                                unsigned short* __restrict__ dst, size_t n) {
  size_t i = (size_t)blockIdx.x * blockDim.x + threadIdx.x;
  size_t stride = (size_t)gridDim.x * blockDim.x;
  for (; i < n; i += stride) dst[i] = f2bf(src[i]);
}

// ---------------- tiled bf16 WMMA GEMM: C[M,N] = A[M,K] * W[N,K]^T ----------------
// block = 256 threads (8 waves), block tile 128x128, k-step 32, LDS staged.
// wave tile 32x64: 2 A-frags x 4 B-frags -> 8 WMMA per k-step per wave.
__launch_bounds__(256)
__global__ void gemm_bf16_kernel(const unsigned short* __restrict__ A,
                                 const unsigned short* __restrict__ W,
                                 float* __restrict__ C, int M, int N, int K) {
  __shared__ __align__(16) unsigned short As[128 * 48];
  __shared__ __align__(16) unsigned short Bs[128 * 48];

  const int tid  = threadIdx.x;
  const int lane = tid & 31;
  const int wv   = tid >> 5;       // wave 0..7
  const int l16  = lane & 15;
  const int half = lane >> 4;

  const int m0 = blockIdx.y * 128;
  const int n0 = blockIdx.x * 128;
  const int mw = (wv & 3) * 32;    // wave row offset  (4 wave-rows)
  const int nw = (wv >> 2) * 64;   // wave col offset  (2 wave-cols)

  const int lrow = tid >> 1;          // 0..127
  const int lcol = (tid & 1) * 16;    // 0 or 16

  FragC c[2][4];
  #pragma unroll
  for (int i = 0; i < 2; ++i)
    #pragma unroll
    for (int j = 0; j < 4; ++j)
      #pragma unroll
      for (int r = 0; r < 8; ++r) c[i][j].f[r] = 0.f;

  const int kbA = half ? 8 : 0;    // unused; A-frags also load at kbB
  const int kbB = half ? 16 : 0;   // B-frag K base per half-wave

  for (int k0 = 0; k0 < K; k0 += 32) {
    __syncthreads();
    const unsigned short* ga = &A[(size_t)(m0 + lrow) * K + k0 + lcol];
    const unsigned short* gw = &W[(size_t)(n0 + lrow) * K + k0 + lcol];
    *(uint4*)&As[lrow * 48 + lcol]     = *(const uint4*)ga;
    *(uint4*)&As[lrow * 48 + lcol + 8] = *(const uint4*)(ga + 8);
    *(uint4*)&Bs[lrow * 48 + lcol]     = *(const uint4*)gw;
    *(uint4*)&Bs[lrow * 48 + lcol + 8] = *(const uint4*)(gw + 8);
    if (k0 + 32 < K) {
      __builtin_prefetch(ga + 32, 0, 0);
      __builtin_prefetch(gw + 32, 0, 0);
    }
    __syncthreads();

    Frag16 af[2], bfv[4];
    #pragma unroll
    for (int i = 0; i < 2; ++i) {
      const unsigned short* ap = &As[(mw + i * 16 + l16) * 48];
      af[i].q[0] = *(const uint4*)(ap + kbB);
      af[i].q[1] = *(const uint4*)(ap + kbB + 8);
    }
    #pragma unroll
    for (int j = 0; j < 4; ++j) {
      const unsigned short* bp = &Bs[(nw + j * 16 + l16) * 48 + kbB];
      bfv[j].q[0] = *(const uint4*)(bp);
      bfv[j].q[1] = *(const uint4*)(bp + 8);
    }
    #pragma unroll
    for (int i = 0; i < 2; ++i)
      #pragma unroll
      for (int j = 0; j < 4; ++j)
        c[i][j].v = __builtin_amdgcn_wmma_f32_16x16x32_bf16(
            false, af[i].v, false, bfv[j].v, (short)0, c[i][j].v, false, false);
  }

  #pragma unroll
  for (int i = 0; i < 2; ++i)
    #pragma unroll
    for (int j = 0; j < 4; ++j)
      #pragma unroll
      for (int r = 0; r < 8; ++r) {
        int mrow = m0 + mw + i * 16 + r + 8 * half;
        C[(size_t)mrow * N + n0 + nw + j * 16 + l16] = c[i][j].f[r];
      }
}

// ---------------- RoPE + layout pack ----------------
// Qf/Kf/Vf: f32 [B,S,DIM] (= [b,s,h,d]); outputs:
//   Qb,Kb : bf16 [B,H,S,D] (rope applied)
//   Vt    : bf16 [B,H,D,S] (transposed, no rope)
__global__ void rope_pack_kernel(const float* __restrict__ Qf,
                                 const float* __restrict__ Kf,
                                 const float* __restrict__ Vf,
                                 const float* __restrict__ fc,
                                 const float* __restrict__ fs,
                                 unsigned short* __restrict__ Qb,
                                 unsigned short* __restrict__ Kb,
                                 unsigned short* __restrict__ Vt) {
  size_t idx = (size_t)blockIdx.x * blockDim.x + threadIdx.x;
  size_t total = (size_t)B_ * S_ * H_ * (D_ / 2);
  if (idx >= total) return;
  int i = idx & 63;                 // pair index 0..63
  size_t t = idx >> 6;
  int h = (int)(t % H_); t /= H_;
  int s = (int)(t % S_);
  int b = (int)(t / S_);

  size_t src = ((size_t)(b * S_ + s) * DIM_) + (size_t)h * D_ + 2 * i;
  float c  = fc[(size_t)s * 64 + i];
  float sn = fs[(size_t)s * 64 + i];

  float qa = Qf[src], qb2 = Qf[src + 1];
  float ka = Kf[src], kb2 = Kf[src + 1];
  float va = Vf[src], vb2 = Vf[src + 1];

  size_t dstq = (((size_t)(b * H_ + h) * S_ + s) * D_) + 2 * i;
  *(unsigned*)&Qb[dstq] = pack2(qa * c - qb2 * sn, qa * sn + qb2 * c);
  *(unsigned*)&Kb[dstq] = pack2(ka * c - kb2 * sn, ka * sn + kb2 * c);

  size_t dstv = ((size_t)(b * H_ + h) * D_ + 2 * i) * S_ + s;
  Vt[dstv]      = f2bf(va);
  Vt[dstv + S_] = f2bf(vb2);
}

// ---------------- flash attention (transposed formulation) ----------------
// One wave per 16-query tile.  S^T = K*Q^T, online softmax per lane column,
// O^T = V^T * P^T.  Output: Ob bf16 [B,S,DIM] (e = h*128 + d).
__launch_bounds__(128)
__global__ void flash_attn_kernel(const unsigned short* __restrict__ Qb,
                                  const unsigned short* __restrict__ Kb,
                                  const unsigned short* __restrict__ Vt,
                                  unsigned short* __restrict__ Ob) {
  const int lane = threadIdx.x & 31;
  const int l16  = lane & 15;
  const int half = lane >> 4;
  const int wid  = blockIdx.x * (blockDim.x >> 5) + (threadIdx.x >> 5);

  const int qt_per = S_ / 16;                 // 128 q-tiles per head
  const int b  = wid / (H_ * qt_per);
  const int rm = wid % (H_ * qt_per);
  const int h  = rm / qt_per;
  const int q0 = (rm % qt_per) * 16;

  const unsigned short* Qh = Qb + ((size_t)(b * H_ + h) * S_) * D_;  // [S][D]
  const unsigned short* Kh = Kb + ((size_t)(b * H_ + h) * S_) * D_;  // [S][D]
  const unsigned short* Vh = Vt + ((size_t)(b * H_ + h) * D_) * S_;  // [D][S]

  const int kbA = half ? 8 : 0;
  const int kbB = half ? 16 : 0;

  // Q as B-fragments for the 4 d-chunks (loaded once)
  Frag16 qf[4];
  #pragma unroll
  for (int dc = 0; dc < 4; ++dc) {
    const unsigned short* p = Qh + (size_t)(q0 + l16) * D_ + dc * 32 + kbB;
    qf[dc].q[0] = *(const uint4*)p;
    qf[dc].q[1] = *(const uint4*)(p + 8);
  }

  FragC acc[8];
  #pragma unroll
  for (int t = 0; t < 8; ++t)
    #pragma unroll
    for (int r = 0; r < 8; ++r) acc[t].f[r] = 0.f;

  float m_run = -1e30f, l_run = 0.f;
  const int q_glob = q0 + l16;
  const int kend = q0 + 16;                   // causal bound

  for (int kc = 0; kc < kend; kc += 32) {
    // ---- S^T tiles: (16 keys x 16 queries) x 2 ----
    FragC st[2];
    #pragma unroll
    for (int tile = 0; tile < 2; ++tile) {
      FragC cc;
      #pragma unroll
      for (int r = 0; r < 8; ++r) cc.f[r] = 0.f;
      #pragma unroll
      for (int dc = 0; dc < 4; ++dc) {
        Frag16 kf;
        const unsigned short* base =
            Kh + (size_t)(kc + tile * 16 + l16) * D_ + dc * 32;
        kf.q[0] = *(const uint4*)(base + kbB);
        kf.q[1] = *(const uint4*)(base + kbB + 8);
        cc.v = __builtin_amdgcn_wmma_f32_16x16x32_bf16(
            false, kf.v, false, qf[dc].v, (short)0, cc.v, false, false);
      }
      st[tile] = cc;
    }

    // ---- scale + causal mask; online softmax (per-lane query) ----
    float cmax = -1e30f;
    #pragma unroll
    for (int tile = 0; tile < 2; ++tile)
      #pragma unroll
      for (int r = 0; r < 8; ++r) {
        int key = kc + tile * 16 + r + 8 * half;
        float s = (key <= q_glob) ? st[tile].f[r] * SCALE_ : -1e30f;
        st[tile].f[r] = s;
        cmax = fmaxf(cmax, s);
      }
    cmax = fmaxf(cmax, __shfl_xor(cmax, 16, 32));
    float m_new = fmaxf(m_run, cmax);
    float alpha = __expf(m_run - m_new);
    float psum = 0.f;
    #pragma unroll
    for (int tile = 0; tile < 2; ++tile)
      #pragma unroll
      for (int r = 0; r < 8; ++r) {
        float p = __expf(st[tile].f[r] - m_new);
        st[tile].f[r] = p;
        psum += p;
      }
    psum += __shfl_xor(psum, 16, 32);
    l_run = l_run * alpha + psum;
    m_run = m_new;
    #pragma unroll
    for (int t = 0; t < 8; ++t)
      #pragma unroll
      for (int r = 0; r < 8; ++r) acc[t].f[r] *= alpha;

    // ---- build P^T B-fragment (32 keys x 16 queries) ----
    float x0[8], x1[8];
    #pragma unroll
    for (int r = 0; r < 8; ++r) {
      x0[r] = __shfl_xor(st[0].f[r], 16, 32);
      x1[r] = __shfl_xor(st[1].f[r], 16, 32);
    }
    Frag16 pt;
    if (half == 0) {
      #pragma unroll
      for (int j = 0; j < 4; ++j)
        pt.u[j] = pack2(st[0].f[2 * j], st[0].f[2 * j + 1]);
      #pragma unroll
      for (int j = 4; j < 8; ++j)
        pt.u[j] = pack2(x0[2 * j - 8], x0[2 * j - 7]);
    } else {
      #pragma unroll
      for (int j = 0; j < 4; ++j)
        pt.u[j] = pack2(x1[2 * j], x1[2 * j + 1]);
      #pragma unroll
      for (int j = 4; j < 8; ++j)
        pt.u[j] = pack2(st[1].f[2 * j - 8], st[1].f[2 * j - 7]);
    }

    // ---- O^T += V^T * P^T  (8 d-tiles) ----
    #pragma unroll
    for (int t = 0; t < 8; ++t) {
      Frag16 vf;
      const unsigned short* base = Vh + (size_t)(t * 16 + l16) * S_ + kc;
      vf.q[0] = *(const uint4*)(base + kbB);
      vf.q[1] = *(const uint4*)(base + kbB + 8);
      acc[t].v = __builtin_amdgcn_wmma_f32_16x16x32_bf16(
          false, vf.v, false, pt.v, (short)0, acc[t].v, false, false);
    }
  }

  // ---- epilogue: normalize, write bf16 O[b, q, h*128 + d] ----
  float inv_l = 1.f / l_run;
  unsigned short* orow = Ob + ((size_t)(b * S_ + q0 + l16)) * DIM_ + h * D_;
  #pragma unroll
  for (int t = 0; t < 8; ++t)
    #pragma unroll
    for (int r = 0; r < 8; r += 2) {
      int d0 = t * 16 + 8 * half + r;
      *(unsigned*)(orow + d0) =
          pack2(acc[t].f[r] * inv_l, acc[t].f[r + 1] * inv_l);
    }
}

// ---------------- adapter K/V projection (tiny, f32) ----------------
__global__ void adapter_kv_kernel(const float* __restrict__ adp,
                                  const float* __restrict__ wk,
                                  const float* __restrict__ wvw,
                                  float* __restrict__ ak,
                                  float* __restrict__ av) {
  int idx = blockIdx.x * blockDim.x + threadIdx.x;
  if (idx >= AL_ * DIM_) return;
  int e = idx % DIM_;
  int l = idx / DIM_;
  const float* arow = adp + (size_t)l * DIM_;
  const float* krow = wk  + (size_t)e * DIM_;
  const float* vrow = wvw + (size_t)e * DIM_;
  float sk = 0.f, sv = 0.f;
  for (int d = 0; d < DIM_; ++d) {
    float a = arow[d];
    sk += a * krow[d];
    sv += a * vrow[d];
  }
  int h = e / D_, dh = e % D_;
  ak[((size_t)h * AL_ + l) * D_ + dh] = sk;
  av[((size_t)h * AL_ + l) * D_ + dh] = sv;
}

// ---------------- gated 10-key adapter attention, added into Ob ----------------
__global__ void adapter_attn_kernel(const unsigned short* __restrict__ Qb,
                                    const float* __restrict__ ak,
                                    const float* __restrict__ av,
                                    const float* __restrict__ gate,
                                    unsigned short* __restrict__ Ob) {
  int idx = blockIdx.x * blockDim.x + threadIdx.x;   // (b,h,s)
  if (idx >= B_ * H_ * S_) return;
  int s = idx % S_;
  int t = idx / S_;
  int h = t % H_;
  int b = t / H_;

  const unsigned short* q = Qb + (((size_t)(b * H_ + h) * S_) + s) * D_;
  float sc[AL_];
  float mx = -1e30f;
  for (int j = 0; j < AL_; ++j) {
    const float* kr = ak + ((size_t)h * AL_ + j) * D_;
    float acc = 0.f;
    for (int d = 0; d < D_; ++d) acc += bf2f(q[d]) * kr[d];
    sc[j] = acc * SCALE_;
    mx = fmaxf(mx, sc[j]);
  }
  float se = 0.f;
  for (int j = 0; j < AL_; ++j) { sc[j] = __expf(sc[j] - mx); se += sc[j]; }
  float inv = 1.f / se;
  float g = tanhf(gate[h]);

  unsigned short* orow = Ob + ((size_t)(b * S_ + s)) * DIM_ + h * D_;
  for (int d = 0; d < D_; ++d) {
    float o = 0.f;
    for (int j = 0; j < AL_; ++j)
      o += sc[j] * av[((size_t)h * AL_ + j) * D_ + d];
    orow[d] = f2bf(bf2f(orow[d]) + g * o * inv);
  }
}

// ---------------- host launcher ----------------
extern "C" void kernel_launch(void* const* d_in, const int* in_sizes, int n_in,
                              void* d_out, int out_size, void* d_ws, size_t ws_size,
                              hipStream_t stream) {
  (void)in_sizes; (void)n_in; (void)out_size; (void)ws_size;
  const float* x       = (const float*)d_in[0];
  const float* wq      = (const float*)d_in[1];
  const float* wk      = (const float*)d_in[2];
  const float* wv      = (const float*)d_in[3];
  const float* wo      = (const float*)d_in[4];
  const float* gate    = (const float*)d_in[5];
  const float* adapter = (const float*)d_in[6];
  const float* fc      = (const float*)d_in[7];
  const float* fs      = (const float*)d_in[8];
  (void)d_in[9];   // mask (causal, implemented in-kernel)
  (void)d_in[10];  // start_pos == 0

  // workspace carve-up (bytes)
  char* ws = (char*)d_ws;
  size_t off = 0;
  auto carve = [&](size_t bytes) -> void* {
    void* p = ws + off;
    off += (bytes + 255) & ~(size_t)255;
    return p;
  };
  const size_t EL = (size_t)MT_ * DIM_;           // 16M elements
  unsigned short* xb  = (unsigned short*)carve(EL * 2);
  unsigned short* wqb = (unsigned short*)carve(EL * 2);
  unsigned short* wkb = (unsigned short*)carve(EL * 2);
  unsigned short* wvb = (unsigned short*)carve(EL * 2);
  unsigned short* wob = (unsigned short*)carve(EL * 2);
  float* Qf = (float*)carve(EL * 4);
  float* Kf = (float*)carve(EL * 4);
  float* Vf = (float*)carve(EL * 4);
  unsigned short* Qb = (unsigned short*)carve(EL * 2);
  unsigned short* Kb = (unsigned short*)carve(EL * 2);
  unsigned short* Vt = (unsigned short*)carve(EL * 2);
  unsigned short* Ob = (unsigned short*)carve(EL * 2);
  float* ak = (float*)carve((size_t)AL_ * DIM_ * 4);
  float* av = (float*)carve((size_t)AL_ * DIM_ * 4);

  // 1) bf16 conversion
  cvt_bf16_kernel<<<2048, 256, 0, stream>>>(x,  xb,  EL);
  cvt_bf16_kernel<<<2048, 256, 0, stream>>>(wq, wqb, EL);
  cvt_bf16_kernel<<<2048, 256, 0, stream>>>(wk, wkb, EL);
  cvt_bf16_kernel<<<2048, 256, 0, stream>>>(wv, wvb, EL);
  cvt_bf16_kernel<<<2048, 256, 0, stream>>>(wo, wob, EL);

  // 2) QKV projections (WMMA GEMM), 128x128 block tiles
  dim3 gg(DIM_ / 128, MT_ / 128);
  gemm_bf16_kernel<<<gg, 256, 0, stream>>>(xb, wqb, Qf, MT_, DIM_, DIM_);
  gemm_bf16_kernel<<<gg, 256, 0, stream>>>(xb, wkb, Kf, MT_, DIM_, DIM_);
  gemm_bf16_kernel<<<gg, 256, 0, stream>>>(xb, wvb, Vf, MT_, DIM_, DIM_);

  // 3) RoPE + repack (+ V transpose)
  {
    size_t total = (size_t)B_ * S_ * H_ * (D_ / 2);
    int blocks = (int)((total + 255) / 256);
    rope_pack_kernel<<<blocks, 256, 0, stream>>>(Qf, Kf, Vf, fc, fs, Qb, Kb, Vt);
  }

  // 4) adapter K/V
  adapter_kv_kernel<<<(AL_ * DIM_ + 255) / 256, 256, 0, stream>>>(adapter, wk, wv, ak, av);

  // 5) causal flash attention (one wave per 16-query tile; 4 waves/block)
  {
    int wave_tiles = B_ * H_ * (S_ / 16);          // 8192
    flash_attn_kernel<<<wave_tiles / 4, 128, 0, stream>>>(Qb, Kb, Vt, Ob);
  }

  // 6) gated adapter attention added into O
  adapter_attn_kernel<<<(B_ * H_ * S_ + 255) / 256, 256, 0, stream>>>(Qb, ak, av, gate, Ob);

  // 7) output projection -> d_out (f32)
  gemm_bf16_kernel<<<gg, 256, 0, stream>>>(Ob, wob, (float*)d_out, MT_, DIM_, DIM_);
}
